// GNN_71657234366696
// MI455X (gfx1250) — matrix operand order repaired and checked by reading, not compiled, not run on hardware
//
#include <hip/hip_runtime.h>
#include <hip/hip_bf16.h>
#include <limits.h>

// ---------------------------------------------------------------------------
// GNN (GAT-like) forward for MI455X / gfx1250.
//  - All dense projections use v_wmma_f32_16x16x32_bf16 (bf16 in, f32 acc).
//    Each wave computes a 16x(16*CT) output strip (CT=4 for the big GEMMs) so
//    the A tile is loaded once per k-step and reused across 4 WMMAs.
//  - Edge gather/scatter kernels are L2-resident (N*256*4B = 30MB << 192MB L2)
//    and use global_atomic_add_f32 for segment sums.
//  - Segment softmax: ordered-int atomicMax pass, then exp + atomicAdd pass.
// ---------------------------------------------------------------------------

#define NN     30000
#define NE     480000
#define NEF    480000
#define DD     128
#define HD2    256
#define GSCALE 0.01f
#define LNEG   0.2f

typedef __attribute__((ext_vector_type(16))) __bf16 v16bf;
typedef __attribute__((ext_vector_type(8)))  float  v8f;

__device__ __forceinline__ float siluf(float v) { return v * (1.0f / (1.0f + __expf(-v))); }

__device__ __forceinline__ __bf16 f2bf(float f) {
  unsigned u = __float_as_uint(f);
  u += 0x7fffu + ((u >> 16) & 1u);           // round-to-nearest-even
  unsigned short s = (unsigned short)(u >> 16);
  return __builtin_bit_cast(__bf16, s);
}
__device__ __forceinline__ int f2ord(float f) {    // monotonic float->int
  int i = __float_as_int(f);
  return (i >= 0) ? i : (i ^ 0x7fffffff);
}
__device__ __forceinline__ float ord2f(int i) {
  return __int_as_float((i >= 0) ? i : (i ^ 0x7fffffff));
}

// --------------------------- utility kernels -------------------------------

__global__ void k_cvt_bf16(const float* __restrict__ s, __bf16* __restrict__ d, int n) {
  int i = blockIdx.x * 256 + threadIdx.x;
  if (i < n) d[i] = f2bf(s[i]);
}
__global__ void k_fill_f32(float* __restrict__ p, float v, int n) {
  int i = blockIdx.x * 256 + threadIdx.x;
  if (i < n) p[i] = v;
}
__global__ void k_fill_i32(int* __restrict__ p, int v, int n) {
  int i = blockIdx.x * 256 + threadIdx.x;
  if (i < n) p[i] = v;
}
__global__ void k_copy_f32(float* __restrict__ d, const float* __restrict__ s, int n) {
  int i = blockIdx.x * 256 + threadIdx.x;
  if (i < n) d[i] = s[i];
}

// x0[n,d] = sum of 5 embedding tables (also emit bf16 copy for the first GEMM)
__global__ void k_embed(const int* __restrict__ atom,
                        const float* __restrict__ e0, const float* __restrict__ e1,
                        const float* __restrict__ e2, const float* __restrict__ e3,
                        const float* __restrict__ e4,
                        float* __restrict__ xf, __bf16* __restrict__ xbf) {
  int tid = blockIdx.x * 256 + threadIdx.x;
  if (tid >= NN * DD) return;
  int n = tid >> 7, d = tid & 127;
  const int* a = atom + n * 5;
  float v = e0[a[0] * DD + d] + e1[a[1] * DD + d] + e2[a[2] * DD + d] +
            e3[a[3] * DD + d] + e4[a[4] * DD + d];
  xf[tid]  = v;
  xbf[tid] = f2bf(v);
}

// LayerNorm over D=128, one wave per row; optional f32 and bf16 outputs.
__global__ void k_layernorm(const float* __restrict__ x, const float* __restrict__ g,
                            const float* __restrict__ b, float* __restrict__ yf,
                            __bf16* __restrict__ ybf, int n) {
  int gid = blockIdx.x * 256 + threadIdx.x;
  int row = gid >> 5, lane = gid & 31;
  if (row >= n) return;
  const float* xr = x + (size_t)row * DD;
  float v[4];
  float s = 0.0f;
#pragma unroll
  for (int i = 0; i < 4; ++i) { v[i] = xr[lane + 32 * i]; s += v[i]; }
#pragma unroll
  for (int o = 16; o > 0; o >>= 1) s += __shfl_xor(s, o, 32);
  float mean = s * (1.0f / DD);
  float q = 0.0f;
#pragma unroll
  for (int i = 0; i < 4; ++i) { float d = v[i] - mean; q += d * d; }
#pragma unroll
  for (int o = 16; o > 0; o >>= 1) q += __shfl_xor(q, o, 32);
  float rstd = rsqrtf(q * (1.0f / DD) + 1e-5f);
#pragma unroll
  for (int i = 0; i < 4; ++i) {
    int col = lane + 32 * i;
    float val = (v[i] - mean) * rstd * g[col] + b[col];
    size_t o = (size_t)row * DD + col;
    if (yf)  yf[o]  = val;
    if (ybf) ybf[o] = f2bf(val);
  }
}

// ------------------------ WMMA bf16 GEMM -----------------------------------
// Y[nrows,M] = act(X[nrows,K] @ W[M,K]^T + bias) (+ residual)
// One wave per 16x(16*CT) output strip, K-loop in steps of 32; the A tile is
// loaded once per k-step and reused across CT WMMAs.
// A layout (16x32 bf16, ISA 7.12.2): lane m=l&15, half=l>>4;
//   vgpr j: K = (j<4?0:16) + half*8 + 2*(j&3), +1
// B layout (32x16 bf16): lane n=l&15, half=l>>4; vgpr j: K = (j<4?0:16) + half*8 + 2*(j&3), +1
// C/D layout: vgpr v -> row = half*8 + v, col = l&15
template <int CT>
__global__ void k_wmma_gemm(const __bf16* __restrict__ X, const __bf16* __restrict__ W,
                            const float* __restrict__ bias,
                            float* __restrict__ Y, __bf16* __restrict__ Ybf,
                            const float* __restrict__ res,
                            int nrows, int K, int M, int act) {
  int gid  = blockIdx.x * 256 + threadIdx.x;
  int wave = gid >> 5;
  int lane = threadIdx.x & 31;
  int groupsM = (M >> 4) / CT;                 // column-tile groups per row
  int tr = wave / groupsM;
  int tg = wave - tr * groupsM;                // column group
  if (tr * 16 >= nrows) return;                // wave-uniform: EXEC all-ones
  int half = lane >> 4;
  int i16  = lane & 15;
  const __bf16* xrow = X + (size_t)(tr * 16 + i16) * K;
  const __bf16* wrow[CT];
#pragma unroll
  for (int ct = 0; ct < CT; ++ct)
    wrow[ct] = W + (size_t)((tg * CT + ct) * 16 + i16) * K;
  v8f c[CT];
#pragma unroll
  for (int ct = 0; ct < CT; ++ct) c[ct] = (v8f){};
  for (int kk = 0; kk < K; kk += 32) {
    v16bf a;
#pragma unroll
    for (int j = 0; j < 8; ++j) {
      int ka = kk + ((j < 4) ? 0 : 16) + half * 8 + 2 * (j & 3);
      a[2 * j]     = xrow[ka];
      a[2 * j + 1] = xrow[ka + 1];
    }
#pragma unroll
    for (int ct = 0; ct < CT; ++ct) {
      v16bf b;
#pragma unroll
      for (int j = 0; j < 8; ++j) {
        int kb = kk + ((j < 4) ? 0 : 16) + half * 8 + 2 * (j & 3);
        b[2 * j]     = wrow[ct][kb];
        b[2 * j + 1] = wrow[ct][kb + 1];
      }
      c[ct] = __builtin_amdgcn_wmma_f32_16x16x32_bf16(false, a, false, b, (short)0,
                                                      c[ct], false, false);
    }
  }
#pragma unroll
  for (int ct = 0; ct < CT; ++ct) {
    int col = (tg * CT + ct) * 16 + i16;
    float bv = bias ? bias[col] : 0.0f;
#pragma unroll
    for (int v = 0; v < 8; ++v) {
      int row = tr * 16 + half * 8 + v;
      float val = c[ct][v] + bv;
      if (act == 1) val = siluf(val);
      size_t o = (size_t)row * M + col;
      if (res) val += res[o];
      if (Y)   Y[o]   = val;
      if (Ybf) Ybf[o] = f2bf(val);
    }
  }
}

// ------------------------- edge kernels ------------------------------------

// local message path: msg = mean_h silu(xs[s,h,:]+xt[t,h,:]+edge_emb[h,:]),
// scatter-added into acc[t,:]. 128 threads per edge (one per d).
__global__ void k_edge_msg(const float* __restrict__ xs, const float* __restrict__ xt,
                           const float* __restrict__ ee1, const float* __restrict__ ee2,
                           const float* __restrict__ ee3,
                           const int* __restrict__ ei0, const int* __restrict__ ei1,
                           const int* __restrict__ ea,
                           float* __restrict__ acc) {
  long long tid = (long long)blockIdx.x * 256 + threadIdx.x;
  int e = (int)(tid >> 7);
  int d = (int)(tid & 127);
  if (e >= NE + NN) return;
  int s, t, a0, a1i, a2i;
  if (e < NE) {
    s = ei0[e]; t = ei1[e];
    a0 = ea[e * 3]; a1i = ea[e * 3 + 1]; a2i = ea[e * 3 + 2];
  } else {                                   // self loops, attr = [25,0,0]
    s = t = e - NE; a0 = 25; a1i = 0; a2i = 0;
  }
  float m = 0.0f;
#pragma unroll
  for (int h = 0; h < 2; ++h) {
    int c = h * DD + d;
    float v = xs[(size_t)s * HD2 + c] + xt[(size_t)t * HD2 + c] +
              ee1[a0 * HD2 + c] + ee2[a1i * HD2 + c] + ee3[a2i * HD2 + c];
    m += siluf(v);
  }
  atomicAdd(&acc[(size_t)t * DD + d], 0.5f * m);
}

// attn projections a1/a2 (D -> H each): out[n, {a1h0,a1h1,a2h0,a2h1}]
__global__ void k_attn_proj(const float* __restrict__ xn,
                            const float* __restrict__ a1W, const float* __restrict__ a1b,
                            const float* __restrict__ a2W, const float* __restrict__ a2b,
                            float* __restrict__ out) {
  int tid = blockIdx.x * 256 + threadIdx.x;
  if (tid >= NN * 4) return;
  int n = tid >> 2, c = tid & 3;
  const float* w = ((c < 2) ? a1W : a2W) + (c & 1) * DD;
  float s = ((c < 2) ? a1b : a2b)[c & 1];
  const float* xr = xn + (size_t)n * DD;
  for (int k = 0; k < DD; ++k) s += xr[k] * w[k];
  out[tid] = s;
}

// per-edge logits = fc_attn(leaky(a1x[s]+a2x[t])); atomicMax per (t,h)
__global__ void k_edge_logits(const float* __restrict__ at4,
                              const int* __restrict__ i0, const int* __restrict__ i1,
                              const float* __restrict__ fcW, const float* __restrict__ fcb,
                              float* __restrict__ logit, int* __restrict__ nmax) {
  int e = blockIdx.x * 256 + threadIdx.x;
  if (e >= NEF) return;
  int s = i0[e], t = i1[e];
  float l0 = at4[s * 4 + 0] + at4[t * 4 + 2];
  float l1 = at4[s * 4 + 1] + at4[t * 4 + 3];
  l0 = (l0 > 0.0f) ? l0 : LNEG * l0;
  l1 = (l1 > 0.0f) ? l1 : LNEG * l1;
  float g0 = fcW[0] * l0 + fcW[1] * l1 + fcb[0];
  float g1 = fcW[2] * l0 + fcW[3] * l1 + fcb[1];
  logit[(size_t)e * 2]     = g0;
  logit[(size_t)e * 2 + 1] = g1;
  atomicMax(&nmax[t * 2],     f2ord(g0));
  atomicMax(&nmax[t * 2 + 1], f2ord(g1));
}

// ex = exp(logit - max[t]); atomicAdd into per-(t,h) sums. In-place over logit.
__global__ void k_edge_exp(float* __restrict__ lg, const int* __restrict__ i1,
                           const int* __restrict__ nmax, float* __restrict__ nsum) {
  int e = blockIdx.x * 256 + threadIdx.x;
  if (e >= NEF) return;
  int t = i1[e];
#pragma unroll
  for (int h = 0; h < 2; ++h) {
    float m = ord2f(nmax[t * 2 + h]);
    float v = __expf(lg[(size_t)e * 2 + h] - m);
    lg[(size_t)e * 2 + h] = v;
    atomicAdd(&nsum[t * 2 + h], v);
  }
}

// full-connect path: one wave per edge. 16 silu values computed by lanes 0..15
// and shared via shfl; each lane expands 4 of the 128 output dims via w5 (RC=8).
__global__ void k_edge_fc(const float* __restrict__ x3, const float* __restrict__ x4,
                          const int* __restrict__ i0, const int* __restrict__ i1,
                          const float* __restrict__ ex, const float* __restrict__ nsum,
                          const float* __restrict__ w5W, const float* __restrict__ w5b,
                          float* __restrict__ acc) {
  int gid = blockIdx.x * 256 + threadIdx.x;
  int e = gid >> 5, lane = threadIdx.x & 31;
  if (e >= NEF) return;
  int s = i0[e], t = i1[e];
  int i = lane & 15;
  float tv = siluf(x3[(size_t)s * 16 + i] + x4[(size_t)t * 16 + i]);
  float tval[16];
#pragma unroll
  for (int j = 0; j < 16; ++j) tval[j] = __shfl(tv, j, 32);
  float a0 = ex[(size_t)e * 2]     / (nsum[t * 2]     + 1e-16f);
  float a1 = ex[(size_t)e * 2 + 1] / (nsum[t * 2 + 1] + 1e-16f);
#pragma unroll
  for (int dd = 0; dd < 4; ++dd) {
    int d = lane + 32 * dd;
    const float* wr = w5W + d * 8;
    float s0 = w5b[d], s1 = w5b[d];
#pragma unroll
    for (int r = 0; r < 8; ++r) { s0 += tval[r] * wr[r]; s1 += tval[8 + r] * wr[r]; }
    atomicAdd(&acc[(size_t)t * DD + d], GSCALE * 0.5f * (a0 * s0 + a1 * s1));
  }
}

// ------------------------------- host --------------------------------------

static inline int cdiv(long long a, long long b) { return (int)((a + b - 1) / b); }

// per-layer param indices (insertion-order flatten of the reference pytree)
enum { J_N1G = 0, J_N1B, J_N2G, J_N2B, J_N3G, J_N3B, J_N4G, J_N4B,
       J_W1W, J_W1B, J_W2W, J_W2B, J_W3W, J_W3B, J_W4W, J_W4B, J_W5W, J_W5B,
       J_A1W, J_A1B, J_A2W, J_A2B, J_FCW, J_FCB, J_EE1, J_EE2, J_EE3,
       J_F1W, J_F1B, J_F2W, J_F2B, J_COUNT };

extern "C" void kernel_launch(void* const* d_in, const int* in_sizes, int n_in,
                              void* d_out, int out_size, void* d_ws, size_t ws_size,
                              hipStream_t stream) {
  (void)in_sizes; (void)n_in; (void)out_size; (void)ws_size;
  const float* embT[5];
  for (int i = 0; i < 5; ++i) embT[i] = (const float*)d_in[i];
  const float* emb1W = (const float*)d_in[5];
  const float* emb1b = (const float*)d_in[6];
  const float* emb2W = (const float*)d_in[7];
  const float* emb2b = (const float*)d_in[8];
  auto LP = [&](int l, int j) { return (const float*)d_in[9 + J_COUNT * l + j]; };
  const int* node_atom = (const int*)d_in[9 + 2 * J_COUNT + 0];
  const int* eidx      = (const int*)d_in[9 + 2 * J_COUNT + 1];  // [2,NE]
  const int* eidxf     = (const int*)d_in[9 + 2 * J_COUNT + 2];  // [2,NEF]
  const int* eattr     = (const int*)d_in[9 + 2 * J_COUNT + 3];  // [NE,3]
  // batch (last input) unused for node-level output

  // ---- workspace carve (≈165 MB) ----
  char* wsp = (char*)d_ws;
  size_t off = 0;
  auto carve = [&](size_t bytes) -> void* {
    void* p = wsp + off;
    off += (bytes + 255) & ~(size_t)255;
    return p;
  };
  float*  x     = (float*)carve((size_t)NN * DD * 4);
  float*  acc   = (float*)carve((size_t)NN * DD * 4);
  float*  xnf   = (float*)carve((size_t)NN * DD * 4);
  __bf16* xnbf  = (__bf16*)carve((size_t)NN * DD * 2);
  __bf16* tmpbf = (__bf16*)carve((size_t)NN * DD * 2);
  float*  xs    = (float*)carve((size_t)NN * HD2 * 4);
  float*  xt    = (float*)carve((size_t)NN * HD2 * 4);
  __bf16* hidbf = (__bf16*)carve((size_t)NN * 512 * 2);
  float*  x3    = (float*)carve((size_t)NN * 16 * 4);
  float*  x4    = (float*)carve((size_t)NN * 16 * 4);
  float*  at4   = (float*)carve((size_t)NN * 4 * 4);
  int*    nmax  = (int*)carve((size_t)NN * 2 * 4);
  float*  nsum  = (float*)carve((size_t)NN * 2 * 4);
  float*  exb   = (float*)carve((size_t)NEF * 2 * 4);
  __bf16* wc    = (__bf16*)carve((size_t)500000 * 2);   // bf16 weight staging

  // bf16 weight slots
  size_t wo = 0;
  auto wslot = [&](size_t n) { __bf16* p = wc + wo; wo += n; return p; };
  __bf16* emb1bf = wslot(128 * 128);
  __bf16* emb2bf = wslot(128 * 128);
  __bf16 *w1bf[2], *w2bf[2], *w3bf[2], *w4bf[2], *f1bf[2], *f2wbf[2];
  for (int l = 0; l < 2; ++l) {
    w1bf[l]  = wslot(256 * 128);
    w2bf[l]  = wslot(256 * 128);
    w3bf[l]  = wslot(16 * 128);
    w4bf[l]  = wslot(16 * 128);
    f1bf[l]  = wslot(512 * 128);
    f2wbf[l] = wslot(128 * 512);
  }

  auto cvt = [&](const float* s, __bf16* d, int n) {
    k_cvt_bf16<<<cdiv(n, 256), 256, 0, stream>>>(s, d, n);
  };
  cvt(emb1W, emb1bf, 128 * 128);
  cvt(emb2W, emb2bf, 128 * 128);
  for (int l = 0; l < 2; ++l) {
    cvt(LP(l, J_W1W), w1bf[l], 256 * 128);
    cvt(LP(l, J_W2W), w2bf[l], 256 * 128);
    cvt(LP(l, J_W3W), w3bf[l], 16 * 128);
    cvt(LP(l, J_W4W), w4bf[l], 16 * 128);
    cvt(LP(l, J_F1W), f1bf[l], 512 * 128);
    cvt(LP(l, J_F2W), f2wbf[l], 128 * 512);
  }

  auto gemm = [&](const __bf16* X, const __bf16* W, const float* bias, float* Y,
                  __bf16* Ybf, const float* res, int K, int M, int act) {
    int tilesM = M / 16;
    if (tilesM % 4 == 0) {
      long long waves = (long long)(NN / 16) * (tilesM / 4);
      k_wmma_gemm<4><<<cdiv(waves * 32, 256), 256, 0, stream>>>(X, W, bias, Y, Ybf,
                                                                res, NN, K, M, act);
    } else {
      long long waves = (long long)(NN / 16) * tilesM;
      k_wmma_gemm<1><<<cdiv(waves * 32, 256), 256, 0, stream>>>(X, W, bias, Y, Ybf,
                                                                res, NN, K, M, act);
    }
  };
  auto lnorm = [&](const float* in, int l, int which_g, int which_b, float* yf,
                   __bf16* ybf) {
    k_layernorm<<<cdiv((long long)NN * 32, 256), 256, 0, stream>>>(
        in, LP(l, which_g), LP(l, which_b), yf, ybf, NN);
  };

  // ---- embedding MLP: x = fc2(silu(fc1(sum_embeds))) ----
  k_embed<<<cdiv((long long)NN * DD, 256), 256, 0, stream>>>(
      node_atom, embT[0], embT[1], embT[2], embT[3], embT[4], xnf, xnbf);
  gemm(xnbf, emb1bf, emb1b, nullptr, tmpbf, nullptr, 128, 128, /*silu*/ 1);
  gemm(tmpbf, emb2bf, emb2b, x, nullptr, nullptr, 128, 128, 0);

  // ---- GAT layers ----
  for (int l = 0; l < 2; ++l) {
    // local message path
    lnorm(x, l, J_N1G, J_N1B, nullptr, xnbf);
    gemm(xnbf, w1bf[l], LP(l, J_W1B), xs, nullptr, nullptr, 128, 256, 0);
    gemm(xnbf, w2bf[l], LP(l, J_W2B), xt, nullptr, nullptr, 128, 256, 0);
    k_copy_f32<<<cdiv((long long)NN * DD, 256), 256, 0, stream>>>(acc, x, NN * DD);
    k_edge_msg<<<cdiv((long long)(NE + NN) * 128, 256), 256, 0, stream>>>(
        xs, xt, LP(l, J_EE1), LP(l, J_EE2), LP(l, J_EE3),
        eidx, eidx + NE, eattr, acc);

    // full-connect path
    lnorm(x, l, J_N2G, J_N2B, nullptr, xnbf);
    gemm(xnbf, w3bf[l], LP(l, J_W3B), x3, nullptr, nullptr, 128, 16, 0);
    gemm(xnbf, w4bf[l], LP(l, J_W4B), x4, nullptr, nullptr, 128, 16, 0);
    lnorm(x, l, J_N3G, J_N3B, xnf, nullptr);
    k_attn_proj<<<cdiv((long long)NN * 4, 256), 256, 0, stream>>>(
        xnf, LP(l, J_A1W), LP(l, J_A1B), LP(l, J_A2W), LP(l, J_A2B), at4);
    k_fill_i32<<<cdiv(NN * 2, 256), 256, 0, stream>>>(nmax, INT_MIN, NN * 2);
    k_fill_f32<<<cdiv(NN * 2, 256), 256, 0, stream>>>(nsum, 0.0f, NN * 2);
    k_edge_logits<<<cdiv(NEF, 256), 256, 0, stream>>>(
        at4, eidxf, eidxf + NEF, LP(l, J_FCW), LP(l, J_FCB), exb, nmax);
    k_edge_exp<<<cdiv(NEF, 256), 256, 0, stream>>>(exb, eidxf + NEF, nmax, nsum);
    k_edge_fc<<<cdiv((long long)NEF * 32, 256), 256, 0, stream>>>(
        x3, x4, eidxf, eidxf + NEF, exb, nsum, LP(l, J_W5W), LP(l, J_W5B), acc);

    // FFN with residual; last layer writes straight to d_out
    lnorm(acc, l, J_N4G, J_N4B, nullptr, xnbf);
    gemm(xnbf, f1bf[l], LP(l, J_F1B), nullptr, hidbf, nullptr, 128, 512, /*silu*/ 1);
    float* yout = (l == 1) ? (float*)d_out : x;
    gemm(hidbf, f2wbf[l], LP(l, J_F2B), yout, nullptr, acc, 512, 128, 0);
  }
}
